// TopkRouter_26448408609432
// MI455X (gfx1250) — matrix operand order repaired and checked by reading, not compiled, not run on hardware
//
#include <hip/hip_runtime.h>
#include <hip/hip_bf16.h>

// ---- CDNA5 (gfx1250) types for WMMA ----
typedef __attribute__((ext_vector_type(16))) __bf16         v16bf;
typedef __attribute__((ext_vector_type(16))) unsigned short v16u;
typedef __attribute__((ext_vector_type(8)))  unsigned int   v8u;
typedef __attribute__((ext_vector_type(8)))  float          v8f;

#define D_DIM        4096   // n_embed
#define E_DIM        64     // num_experts
#define KTOP         2
#define ROWS_PER_WAVE 16
#define WAVES_PER_BLOCK 4
#define ROWS_PER_BLOCK (ROWS_PER_WAVE * WAVES_PER_BLOCK)   // 64
#define KCHUNK       128    // K staged to LDS per round
#define KSUB         32     // K per v_wmma_f32_16x16x32_bf16

// Pack two fp32 into two bf16 (truncation) with a single v_perm_b32.
// dst[15:0] = hi16(a), dst[31:16] = hi16(b)
__device__ __forceinline__ unsigned int pack_bf16_trunc(float a, float b) {
    return __builtin_amdgcn_perm(__float_as_uint(b), __float_as_uint(a), 0x07060302u);
}

__global__ __launch_bounds__(WAVES_PER_BLOCK * 32)
void TopkRouter_kernel(const float* __restrict__ x,
                       const float* __restrict__ W,
                       const float* __restrict__ bias,
                       float* __restrict__ out,
                       int nRows) {
    // W chunk staged as bf16 bits: [expert][k] with k in [0,KCHUNK)
    __shared__ __align__(32) unsigned short ldsW[E_DIM * KCHUNK];       // 16 KB
    // logits scratch, padded to 65 to avoid bank conflicts on column scans
    __shared__ float ldsC[ROWS_PER_BLOCK][E_DIM + 1];                   // ~16.25 KB

    const int tid      = threadIdx.x;
    const int lane     = tid & 31;
    const int wave     = tid >> 5;
    const int lane15   = lane & 15;
    const int laneHalf = lane >> 4;          // 0: lanes 0-15, 1: lanes 16-31
    const int blockRow = blockIdx.x * ROWS_PER_BLOCK;

    // A-fragment source row for this lane (lanes L and L+16 share row L, split K)
    const int rowA = blockRow + wave * ROWS_PER_WAVE + lane15;
    const float* xrow = x + (size_t)rowA * D_DIM + laneHalf * 16;

    // Per-lane LDS base for B fragments: lane n of tile t reads expert t*16+n,
    // 16 contiguous bf16 starting at k-substep offset (+16 for lanes 16-31).
    const unsigned short* ldsWlane = &ldsW[lane15 * KCHUNK + laneHalf * 16];

    v8f acc[4];
    const v8f vzero = {0.f, 0.f, 0.f, 0.f, 0.f, 0.f, 0.f, 0.f};
    acc[0] = vzero; acc[1] = vzero; acc[2] = vzero; acc[3] = vzero;

    for (int kc = 0; kc < D_DIM; kc += KCHUNK) {
        __syncthreads();
        // ---- stage W[0:64, kc:kc+128] fp32 -> bf16 LDS; 2048 float4 over 128 threads ----
        #pragma unroll
        for (int i = 0; i < 16; ++i) {
            int q  = tid + i * (WAVES_PER_BLOCK * 32);   // float4 index
            int e  = q >> 5;                             // 32 float4 per expert row
            int k4 = q & 31;
            float4 wv = *(const float4*)(W + (size_t)e * D_DIM + kc + k4 * 4);
            unsigned int p0 = pack_bf16_trunc(wv.x, wv.y);
            unsigned int p1 = pack_bf16_trunc(wv.z, wv.w);
            *(uint2*)&ldsW[e * KCHUNK + k4 * 4] = make_uint2(p0, p1);
        }
        __syncthreads();

        // ---- 4 WMMA K-substeps over the staged chunk ----
        #pragma unroll
        for (int s = 0; s < KCHUNK / KSUB; ++s) {
            const int ks = kc + s * KSUB;

            // A fragment: this lane's row, K = base+{0..15} (base = laneHalf*16)
            const float4* ap = (const float4*)(xrow + ks);
            float4 a0 = ap[0];   // k+0..3
            float4 a1 = ap[1];   // k+4..7
            float4 a2 = ap[2];   // k+8..11
            float4 a3 = ap[3];   // k+12..15

            // B fragments: load ALL 4 tiles up front into distinct registers so the
            // compiler can clause the 8 ds_load_b128, wait once, then issue the 4
            // WMMAs back-to-back (no per-tile LDS round-trip on the critical path).
            v16u braw[4];
            #pragma unroll
            for (int t = 0; t < 4; ++t)
                braw[t] = *(const v16u*)(ldsWlane + t * 16 * KCHUNK + s * KSUB);

            v8u aw;
            aw[0] = pack_bf16_trunc(a0.x, a0.y);
            aw[1] = pack_bf16_trunc(a0.z, a0.w);
            aw[2] = pack_bf16_trunc(a1.x, a1.y);
            aw[3] = pack_bf16_trunc(a1.z, a1.w);
            aw[4] = pack_bf16_trunc(a2.x, a2.y);
            aw[5] = pack_bf16_trunc(a2.z, a2.w);
            aw[6] = pack_bf16_trunc(a3.x, a3.y);
            aw[7] = pack_bf16_trunc(a3.z, a3.w);
            v16bf afrag = __builtin_bit_cast(v16bf, aw);

            #pragma unroll
            for (int t = 0; t < 4; ++t) {
                v16bf bfrag = __builtin_bit_cast(v16bf, braw[t]);
                acc[t] = __builtin_amdgcn_wmma_f32_16x16x32_bf16(
                    false, afrag, false, bfrag, (short)0, acc[t], false, false);
            }
        }
    }

    // ---- scatter accumulators (+bias) to LDS: C tile (m,n) lives at
    // VGPR i = m%8, lane = (m/8)*16 + n  ----
    __syncthreads();
    {
        const int rbase = wave * ROWS_PER_WAVE + laneHalf * 8;
        #pragma unroll
        for (int t = 0; t < 4; ++t) {
            const int col = t * 16 + lane15;
            const float bb = bias[col];
            #pragma unroll
            for (int i = 0; i < 8; ++i)
                ldsC[rbase + i][col] = acc[t][i] + bb;
        }
    }
    __syncthreads();

    // ---- top-2 + softmax: one thread per row ----
    if (tid < ROWS_PER_BLOCK) {
        const int r = tid;
        float v0 = -3.402823466e+38f, v1 = -3.402823466e+38f;
        int   i0 = 0, i1 = 0;
        #pragma unroll 8
        for (int e = 0; e < E_DIM; ++e) {
            float v = ldsC[r][e];
            if (v > v0)      { v1 = v0; i1 = i0; v0 = v; i0 = e; }
            else if (v > v1) { v1 = v;  i1 = e; }
        }
        // softmax over [v0, v1] (v0 is the max); strict '>' above keeps the
        // lowest index on ties, matching jax.lax.top_k ordering.
        float e1  = __expf(v1 - v0);
        float inv = 1.0f / (1.0f + e1);
        const size_t grow = (size_t)(blockRow + r);
        out[grow * KTOP + 0] = (float)i0;                 // indices (as float output dtype)
        out[grow * KTOP + 1] = (float)i1;
        float* gates = out + (size_t)nRows * KTOP;
        gates[grow * KTOP + 0] = inv;
        gates[grow * KTOP + 1] = e1 * inv;
    }
}

extern "C" void kernel_launch(void* const* d_in, const int* in_sizes, int n_in,
                              void* d_out, int out_size, void* d_ws, size_t ws_size,
                              hipStream_t stream) {
    const float* x    = (const float*)d_in[0];   // [B,S,D]
    const float* W    = (const float*)d_in[1];   // [E,D]
    const float* bias = (const float*)d_in[2];   // [E]
    float* out = (float*)d_out;

    const int nRows = in_sizes[0] / D_DIM;       // B*S = 16384
    const int grid  = nRows / ROWS_PER_BLOCK;    // 256 blocks
    TopkRouter_kernel<<<dim3(grid), dim3(WAVES_PER_BLOCK * 32), 0, stream>>>(
        x, W, bias, out, nRows);
}
